// DualSlidingWindowAttention_52312701665774
// MI455X (gfx1250) — matrix operand order repaired and checked by reading, not compiled, not run on hardware
//
#include <hip/hip_runtime.h>
#include <hip/hip_bf16.h>
#include <math.h>

#define HID 2048
#define NH   32
#define NKV   8
#define GQ    4
#define HD   64
#define TT 1024
#define WATT 256
#define WSSM  64
#define KVW (NKV*HD)   // 512
#define BT  (2*TT)     // 2048 rows, batch-major

typedef __attribute__((ext_vector_type(16))) __bf16        v16bf;
typedef __attribute__((ext_vector_type(8)))  float         v8f;
typedef __attribute__((ext_vector_type(4)))  unsigned int  v4u;
typedef __attribute__((ext_vector_type(2)))  float         v2f;
typedef unsigned short u16;

union Frag16 { v16bf v; u16 u[16]; v4u q2[2]; };

static __device__ __forceinline__ u16 f2bf(float f) {
  unsigned u = __float_as_uint(f);
  u += 0x7FFFu + ((u >> 16) & 1u);      // round-to-nearest-even
  return (u16)(u >> 16);
}

static __device__ __forceinline__ v8f vzero8() {
  v8f z;
#pragma unroll
  for (int i = 0; i < 8; ++i) z[i] = 0.0f;
  return z;
}
static __device__ __forceinline__ v4u vzero4u() {
  v4u z;
#pragma unroll
  for (int i = 0; i < 4; ++i) z[i] = 0u;
  return z;
}

static __device__ __forceinline__ v8f wmma_bf16(v16bf a, v16bf b, v8f c) {
  return __builtin_amdgcn_wmma_f32_16x16x32_bf16(false, a, false, b, (short)0, c,
                                                 false, false);
}

static __device__ __forceinline__ v4u ld4(const u16* base, int elt_off) {
  return *reinterpret_cast<const v4u*>(base + elt_off);
}

// ---- DPP width-16 butterfly reductions (pure VALU, no LDS) ------------------
// quad_perm xor1 = 0xB1, quad_perm xor2 = 0x4E, row_half_mirror = 0x141,
// row_mirror = 0x140.  DPP rows are 16 lanes -> reduction confined per half.
static __device__ __forceinline__ float redmax16(float x) {
  x = fmaxf(x, __int_as_float(__builtin_amdgcn_update_dpp(
        0, __float_as_int(x), 0xB1, 0xf, 0xf, true)));
  x = fmaxf(x, __int_as_float(__builtin_amdgcn_update_dpp(
        0, __float_as_int(x), 0x4E, 0xf, 0xf, true)));
  x = fmaxf(x, __int_as_float(__builtin_amdgcn_update_dpp(
        0, __float_as_int(x), 0x141, 0xf, 0xf, true)));
  x = fmaxf(x, __int_as_float(__builtin_amdgcn_update_dpp(
        0, __float_as_int(x), 0x140, 0xf, 0xf, true)));
  return x;
}
static __device__ __forceinline__ float redsum16(float x) {
  x += __int_as_float(__builtin_amdgcn_update_dpp(
        0, __float_as_int(x), 0xB1, 0xf, 0xf, true));
  x += __int_as_float(__builtin_amdgcn_update_dpp(
        0, __float_as_int(x), 0x4E, 0xf, 0xf, true));
  x += __int_as_float(__builtin_amdgcn_update_dpp(
        0, __float_as_int(x), 0x141, 0xf, 0xf, true));
  x += __int_as_float(__builtin_amdgcn_update_dpp(
        0, __float_as_int(x), 0x140, 0xf, 0xf, true));
  return x;
}

// ---------------------------------------------------------------------------
// f32 -> bf16 elementwise (packed pair stores)
// ---------------------------------------------------------------------------
__global__ __launch_bounds__(256)
void cvt_bf16(const float* __restrict__ x, u16* __restrict__ y, int npairs)
{
  const int i = blockIdx.x * 256 + threadIdx.x;
  if (i < npairs) {
    const v2f f = *reinterpret_cast<const v2f*>(x + 2 * (size_t)i);
    const unsigned lo = f2bf(f[0]);
    const unsigned hi = f2bf(f[1]);
    reinterpret_cast<unsigned*>(y)[i] = lo | (hi << 16);
  }
}

// ---------------------------------------------------------------------------
// f32 [K x N] -> bf16 [N x K] tiled transpose (coalesced both sides via LDS)
// grid (N/32, K/32), block (32, 8)
// ---------------------------------------------------------------------------
__global__ __launch_bounds__(256)
void transpose_bf16(const float* __restrict__ W, u16* __restrict__ Wt,
                    int K, int N)
{
  __shared__ float t[32][33];
  const int tx = threadIdx.x, ty = threadIdx.y;
  const int n0 = blockIdx.x * 32, k0 = blockIdx.y * 32;
#pragma unroll
  for (int i = 0; i < 4; ++i)
    t[ty + i * 8][tx] = W[(size_t)(k0 + ty + i * 8) * N + n0 + tx];
  __syncthreads();
#pragma unroll
  for (int i = 0; i < 4; ++i)
    Wt[(size_t)(n0 + ty + i * 8) * K + k0 + tx] = f2bf(t[tx][ty + i * 8]);
}

// ---------------------------------------------------------------------------
// GEMM: C[M x N] = concat(A0,A1)[M x K] @ B[K x N]
// A row-major bf16; B pre-transposed bf16 (Bt[n][k], ldbt = K).
// Block = 8 waves (4 x 2); wave owns 32x64 (2x4 frags, 8 WMMA / k-step).
// MODE: 0 = bf16 row-major out, 1 = bf16 transposed out, 2 = f32 row-major out
// grid (N/128, M/128).
// ---------------------------------------------------------------------------
template <int MODE>
__global__ __launch_bounds__(256)
void gemm_bf16wmma(const u16* __restrict__ A0, int K0,
                   const u16* __restrict__ A1, int K1, int lda,
                   const u16* __restrict__ Bt, int ldbt,
                   void* __restrict__ Cv, int ldc)
{
  const int wave = threadIdx.x >> 5;
  const int lane = threadIdx.x & 31;
  const int r    = lane & 15;
  const int half = lane >> 4;
  const int row0 = blockIdx.y * 128 + (wave & 3) * 32;
  const int col0 = blockIdx.x * 128 + (wave >> 2) * 64;

  v8f acc[2][4];
#pragma unroll
  for (int mi = 0; mi < 2; ++mi)
#pragma unroll
    for (int ni = 0; ni < 4; ++ni) acc[mi][ni] = vzero8();

  const int oa0 = (row0 + r) * lda + half * 8;
  const int oa1 = (row0 + 16 + r) * lda + half * 8;
  int ob[4];
#pragma unroll
  for (int ni = 0; ni < 4; ++ni)
    ob[ni] = (col0 + ni * 16 + r) * ldbt + half * 8;

  int kglobal = 0;   // running k for B (continues across A segments)
#pragma unroll 1
  for (int sgi = 0; sgi < 2; ++sgi) {
    const u16* __restrict__ Ab = sgi ? A1 : A0;
    const int Kc = sgi ? K1 : K0;
#pragma unroll 1
    for (int kk = 0; kk < Kc; kk += 32) {
      Frag16 a[2], b[4];
      a[0].q2[0] = ld4(Ab, oa0 + kk);
      a[0].q2[1] = ld4(Ab, oa0 + kk + 16);
      a[1].q2[0] = ld4(Ab, oa1 + kk);
      a[1].q2[1] = ld4(Ab, oa1 + kk + 16);
      const int kb = kglobal + kk;
#pragma unroll
      for (int ni = 0; ni < 4; ++ni) {
        b[ni].q2[0] = ld4(Bt, ob[ni] + kb);
        b[ni].q2[1] = ld4(Bt, ob[ni] + kb + 16);
      }
#pragma unroll
      for (int mi = 0; mi < 2; ++mi)
#pragma unroll
        for (int ni = 0; ni < 4; ++ni)
          acc[mi][ni] = wmma_bf16(a[mi].v, b[ni].v, acc[mi][ni]);
    }
    kglobal += Kc;
  }

  // ---- branch-free epilogue, one hoisted per-lane base offset ----
  const int rowL = row0 + 8 * half;    // + mi*16 + j
  const int colL = col0 + r;           // + ni*16
  if (MODE == 2) {
    float* __restrict__ Cf = (float*)Cv;
    const int base = rowL * ldc + colL;
#pragma unroll
    for (int mi = 0; mi < 2; ++mi)
#pragma unroll
      for (int ni = 0; ni < 4; ++ni)
#pragma unroll
        for (int j = 0; j < 8; ++j)
          Cf[base + (mi * 16 + j) * ldc + ni * 16] = acc[mi][ni][j];
  } else if (MODE == 1) {
    u16* __restrict__ C = (u16*)Cv;
    const int base = colL * ldc + rowL;
#pragma unroll
    for (int mi = 0; mi < 2; ++mi)
#pragma unroll
      for (int ni = 0; ni < 4; ++ni)
#pragma unroll
        for (int j = 0; j < 8; ++j)
          C[base + (ni * 16) * ldc + mi * 16 + j] = f2bf(acc[mi][ni][j]);
  } else {
    u16* __restrict__ C = (u16*)Cv;
    const int base = rowL * ldc + colL;
#pragma unroll
    for (int mi = 0; mi < 2; ++mi)
#pragma unroll
      for (int ni = 0; ni < 4; ++ni)
#pragma unroll
        for (int j = 0; j < 8; ++j)
          C[base + (mi * 16 + j) * ldc + ni * 16] = f2bf(acc[mi][ni][j]);
  }
}

// ---------------------------------------------------------------------------
// Dual sliding-window attention, flash-style, one wave per (b, h, q-tile16),
// 32 keys per iteration (full-K PV), DPP softmax reductions.
// Q/K/SK bf16 row-major; V/SV bf16 transposed [d][t]; O bf16 row-major.
// ---------------------------------------------------------------------------
__global__ __launch_bounds__(32)
void dswa_attn(const u16* __restrict__ Q,
               const u16* __restrict__ Kb,  const u16* __restrict__ Vt,
               const u16* __restrict__ SKb, const u16* __restrict__ SVt,
               u16* __restrict__ Ob)
{
  const int QTILES = TT / 16;
  const int bid = blockIdx.x;
  const int qt = bid % QTILES;
  const int h  = (bid / QTILES) % NH;
  const int b  =  bid / (QTILES * NH);
  const int kh = h / GQ;
  const int lane = threadIdx.x & 31;
  const int r    = lane & 15;
  const int half = lane >> 4;
  const int q0   = qt * 16;
  const float slope = exp2f(-0.25f * (float)h);   // 2^(-8h/32)

  // Q A-frags (unscaled; 1/sqrt(D) folded into the f32 scores)
  Frag16 aQ[2];
  const int qoff = (b * TT + q0 + r) * HID + h * HD;
#pragma unroll
  for (int c = 0; c < 2; ++c) {
    aQ[c].q2[0] = ld4(Q, qoff + c * 32 + half * 16);
    aQ[c].q2[1] = ld4(Q, qoff + c * 32 + half * 16 + 8);
  }

  float m[8], l[8];
  v8f o[4];
#pragma unroll
  for (int j = 0; j < 8; ++j) { m[j] = -1e30f; l[j] = 0.0f; }
#pragma unroll
  for (int dt = 0; dt < 4; ++dt) o[dt] = vzero8();

  __shared__ float sP[16 * 33];   // 16 rows x 32 cols (+pad)

#pragma unroll 1
  for (int seg = 0; seg < 2; ++seg) {
    const u16* __restrict__ Ks = seg ? SKb : Kb;
    const u16* __restrict__ Vs = seg ? SVt : Vt;
    const int W = seg ? WSSM : WATT;
    int kb0 = q0 - W + 1;
    if (kb0 < 0) kb0 = 0;
    kb0 &= ~31;

#pragma unroll 1
    for (int kbase = kb0; kbase <= q0; kbase += 32) {
      // upper 16 keys contain any valid key? (uniform; both 16-aligned)
      const bool hasHi = (kbase + 16 <= q0);

      // ---- S = Q K^T : 16x32 scores, 2(+2) WMMAs over D=64 ----
      v8f s0 = vzero8(), s1 = vzero8();
      const int ko0 = (b * TT + kbase + r) * KVW + kh * HD;
#pragma unroll
      for (int c = 0; c < 2; ++c) {
        Frag16 bK;
        bK.q2[0] = ld4(Ks, ko0 + c * 32 + half * 16);
        bK.q2[1] = ld4(Ks, ko0 + c * 32 + half * 16 + 8);
        s0 = wmma_bf16(aQ[c].v, bK.v, s0);
      }
      if (hasHi) {
        const int ko1 = ko0 + 16 * KVW;
#pragma unroll
        for (int c = 0; c < 2; ++c) {
          Frag16 bK;
          bK.q2[0] = ld4(Ks, ko1 + c * 32 + half * 16);
          bK.q2[1] = ld4(Ks, ko1 + c * 32 + half * 16 + 8);
          s1 = wmma_bf16(aQ[c].v, bK.v, s1);
        }
      }

      // ---- scale + ALiBi + window mask + online softmax (DPP) ----
      float p0[8], p1[8], alpha[8];
#pragma unroll
      for (int j = 0; j < 8; ++j) {
        const int qq = q0 + j + 8 * half;
        const int d0 = qq - (kbase + r);
        const int d1 = d0 - 16;
        const bool in0 = (d0 >= 0) && (d0 < W);
        const bool in1 = hasHi && (d1 >= 0) && (d1 < W);
        const float sc0 = s0[j] * 0.125f - slope * (float)d0;
        const float sc1 = s1[j] * 0.125f - slope * (float)d1;
        float tv = fmaxf(in0 ? sc0 : -1e30f, in1 ? sc1 : -1e30f);
        tv = redmax16(tv);
        const float newm = fmaxf(m[j], tv);
        const float a = __expf(m[j] - newm);
        const float pe0 = in0 ? __expf(sc0 - newm) : 0.0f;
        const float pe1 = in1 ? __expf(sc1 - newm) : 0.0f;
        const float ps = redsum16(pe0 + pe1);
        l[j] = l[j] * a + ps;
        m[j] = newm;
        p0[j] = pe0;
        p1[j] = pe1;
        alpha[j] = a;
      }
#pragma unroll
      for (int dt = 0; dt < 4; ++dt)
#pragma unroll
        for (int j = 0; j < 8; ++j) o[dt][j] *= alpha[j];

      // ---- transpose P (C layout -> A layout) through LDS ----
      __syncthreads();
#pragma unroll
      for (int j = 0; j < 8; ++j) {
        sP[(j + 8 * half) * 33 + r]      = p0[j];
        sP[(j + 8 * half) * 33 + r + 16] = p1[j];
      }
      __syncthreads();
      Frag16 aP;
#pragma unroll
      for (int i = 0; i < 8; ++i) {
        aP.u[i]     = f2bf(sP[r * 33 + half * 16 + i]);       // K 16h+0..7
        aP.u[8 + i] = f2bf(sP[r * 33 + half * 16 + 8 + i]);   // K 16h+8..15
      }
      __syncthreads();

      // ---- O += P V : 4 WMMAs, full K=32 ----
#pragma unroll
      for (int dt = 0; dt < 4; ++dt) {
        Frag16 bV;
        if (hasHi || half == 0) {
          const int voff = (kh * HD + dt * 16 + r) * BT + b * TT + kbase +
                           16 * half;
          bV.q2[0] = ld4(Vs, voff);        // keys +0..7
          bV.q2[1] = ld4(Vs, voff + 8);    // keys +8..15
        } else {
          bV.q2[0] = vzero4u();            // tail: upper keys all after the last query
          bV.q2[1] = vzero4u();
        }
        o[dt] = wmma_bf16(aP.v, bV.v, o[dt]);
      }
    }
  }

  // ---- normalize and scatter (bf16) ----
#pragma unroll
  for (int dt = 0; dt < 4; ++dt)
#pragma unroll
    for (int j = 0; j < 8; ++j) {
      const int qq = q0 + j + 8 * half;
      Ob[(b * TT + qq) * HID + h * HD + dt * 16 + r] = f2bf(o[dt][j] / l[j]);
    }
}

// ---------------------------------------------------------------------------
extern "C" void kernel_launch(void* const* d_in, const int* in_sizes, int n_in,
                              void* d_out, int out_size, void* d_ws, size_t ws_size,
                              hipStream_t stream) {
  const float* hidden = (const float*)d_in[0];
  const float* ssm    = (const float*)d_in[1];
  const float* Wq  = (const float*)d_in[2];
  const float* Wk  = (const float*)d_in[3];
  const float* Wv  = (const float*)d_in[4];
  const float* Wsk = (const float*)d_in[5];
  const float* Wsv = (const float*)d_in[6];
  const float* Wo  = (const float*)d_in[7];
  float* out = (float*)d_out;

  u16* ws = (u16*)d_ws;
  u16* hb   = ws;                       // hidden bf16      [BT x HID]
  u16* sb   = hb   + (size_t)BT * HID;  // ssm bf16         [BT x HID]
  u16* WqT  = sb   + (size_t)BT * HID;  // Wq^T             [HID x 2HID]
  u16* WkT  = WqT  + (size_t)HID * 2 * HID;   // [KVW x HID]
  u16* WvT  = WkT  + (size_t)KVW * HID;
  u16* WskT = WvT  + (size_t)KVW * HID;
  u16* WsvT = WskT + (size_t)KVW * HID;
  u16* WoT  = WsvT + (size_t)KVW * HID;       // [HID x HID]
  u16* Qb   = WoT  + (size_t)HID * HID;       // [BT x HID]
  u16* Kb   = Qb   + (size_t)BT * HID;        // [BT x KVW]
  u16* SKb  = Kb   + (size_t)BT * KVW;        // [BT x KVW]
  u16* Vtb  = SKb  + (size_t)BT * KVW;        // [KVW x BT] (transposed)
  u16* SVtb = Vtb  + (size_t)KVW * BT;        // [KVW x BT] (transposed)
  u16* Ob   = SVtb + (size_t)KVW * BT;        // [BT x HID]

  const dim3 b256(256);
  const dim3 bT(32, 8);

  // --- one-time precision conversion / weight transposes ---
  cvt_bf16<<<dim3((BT * HID) / 512), b256, 0, stream>>>(hidden, hb, (BT * HID) / 2);
  cvt_bf16<<<dim3((BT * HID) / 512), b256, 0, stream>>>(ssm,    sb, (BT * HID) / 2);
  transpose_bf16<<<dim3(HID / 32, (2 * HID) / 32), bT, 0, stream>>>(Wq,  WqT,  2 * HID, HID);
  transpose_bf16<<<dim3(KVW / 32, HID / 32),       bT, 0, stream>>>(Wk,  WkT,  HID, KVW);
  transpose_bf16<<<dim3(KVW / 32, HID / 32),       bT, 0, stream>>>(Wv,  WvT,  HID, KVW);
  transpose_bf16<<<dim3(KVW / 32, HID / 32),       bT, 0, stream>>>(Wsk, WskT, HID, KVW);
  transpose_bf16<<<dim3(KVW / 32, HID / 32),       bT, 0, stream>>>(Wsv, WsvT, HID, KVW);
  transpose_bf16<<<dim3(HID / 32, HID / 32),       bT, 0, stream>>>(Wo,  WoT,  HID, HID);

  // --- projections (bf16 WMMA) ---
  // Q = [ssm | hidden] @ Wq   (M=2048, K=4096, N=2048)
  gemm_bf16wmma<0><<<dim3(HID / 128, BT / 128), b256, 0, stream>>>(
      sb, HID, hb, HID, HID, WqT, 2 * HID, Qb, HID);
  // K = hidden @ Wk ; SK = ssm @ Wsk  (row-major bf16)
  gemm_bf16wmma<0><<<dim3(KVW / 128, BT / 128), b256, 0, stream>>>(
      hb, HID, hb, 0, HID, WkT, HID, Kb, KVW);
  gemm_bf16wmma<0><<<dim3(KVW / 128, BT / 128), b256, 0, stream>>>(
      sb, HID, sb, 0, HID, WskT, HID, SKb, KVW);
  // V = hidden @ Wv ; SV = ssm @ Wsv  (stored transposed [d][t])
  gemm_bf16wmma<1><<<dim3(KVW / 128, BT / 128), b256, 0, stream>>>(
      hb, HID, hb, 0, HID, WvT, HID, Vtb, BT);
  gemm_bf16wmma<1><<<dim3(KVW / 128, BT / 128), b256, 0, stream>>>(
      sb, HID, sb, 0, HID, WsvT, HID, SVtb, BT);

  // --- windowed dual attention ---
  dswa_attn<<<dim3(2 * NH * (TT / 16)), dim3(32), 0, stream>>>(
      Qb, Kb, Vtb, SKb, SVtb, Ob);

  // --- out = O @ Wo (f32 output) ---
  gemm_bf16wmma<2><<<dim3(HID / 128, BT / 128), b256, 0, stream>>>(
      Ob, HID, Ob, 0, HID, WoT, HID, out, HID);
}
